// MultipleEmbedding_34316788695560
// MI455X (gfx1250) — matrix operand order repaired and checked by reading, not compiled, not run on hardware
//
#include <hip/hip_runtime.h>
#include <hip/hip_bf16.h>
#include <cstdint>

typedef __bf16 bf16;
typedef __attribute__((ext_vector_type(4)))  __bf16 v4bf;
typedef __attribute__((ext_vector_type(8)))  __bf16 v8bf;
typedef __attribute__((ext_vector_type(16))) __bf16 v16bf;
typedef __attribute__((ext_vector_type(8)))  float  v8f;

#define BATCH 4096
#define NN    8192
#define DD    1024
#define S1    72          // GEMM1 LDS row stride in halves (144B, 16B-aligned, skewed)
#define S2    40          // GEMM2 LDS row stride in halves (80B, 16B-aligned, skewed)
#define INV_KEEP (1.0f / 0.75f)

// ---------------------------------------------------------------- helpers
__device__ __forceinline__ v16bf ld16(const bf16* p0, const bf16* p1) {
  v8bf lo = *(const v8bf*)p0;
  v8bf hi = *(const v8bf*)p1;
  v16bf r;
#pragma unroll
  for (int i = 0; i < 8; ++i) { r[i] = lo[i]; r[i + 8] = hi[i]; }
  return r;
}

// branch-free tanh: 1 - 2/(e^{2x}+1); exact at +-inf, no EXEC divergence
__device__ __forceinline__ float fast_tanh(float y) {
  float e = __expf(2.0f * y);
  return 1.0f - 2.0f / (e + 1.0f);
}

// one 16x16x32 MMA stage for a 2x4-fragment wave tile, from LDS tiles
template <int STRIDE>
__device__ __forceinline__ void mma_stage(const bf16* __restrict__ As,
                                          const bf16* __restrict__ Bs, int kb,
                                          v8f (&acc)[2][4], int waveM, int waveN, int lane) {
  const int aoff = (lane & 16) ? 8 : 0;
  const int boff = (lane & 16) ? 8 : 0;
  const int arow = lane & 15;
  v16bf af[2], bfr[4];
#pragma unroll
  for (int fi = 0; fi < 2; ++fi) {
    const bf16* p = As + (waveM * 32 + fi * 16 + arow) * STRIDE + kb;
    af[fi] = ld16(p + aoff, p + aoff + 16);
  }
#pragma unroll
  for (int fj = 0; fj < 4; ++fj) {
    const bf16* p = Bs + (waveN * 64 + fj * 16 + arow) * STRIDE + kb;
    bfr[fj] = ld16(p + boff, p + boff + 16);
  }
#pragma unroll
  for (int fi = 0; fi < 2; ++fi)
#pragma unroll
    for (int fj = 0; fj < 4; ++fj)
      acc[fi][fj] = __builtin_amdgcn_wmma_f32_16x16x32_bf16(
          false, af[fi], false, bfr[fj], (short)0, acc[fi][fj], false, false);
}

// async global->LDS 16B copy (ASYNCcnt-tracked, bypasses VGPRs)
__device__ __forceinline__ void async_cp16(uint32_t lds_addr, uint32_t byte_off, uint64_t base) {
  asm volatile("global_load_async_to_lds_b128 %0, %1, %2 offset:0"
               :: "v"(lds_addr), "v"(byte_off), "s"(base) : "memory");
}
__device__ __forceinline__ void wait_async_le4() {
  asm volatile("s_wait_asynccnt 0x4" ::: "memory");
}
__device__ __forceinline__ void wait_async_0() {
  asm volatile("s_wait_asynccnt 0x0" ::: "memory");
}

// ---------------------------------------------------------------- k0: zero accumulators
__global__ void zero_kernel(float* __restrict__ a, int n) {
  int i = blockIdx.x * blockDim.x + threadIdx.x;
  if (i < n) a[i] = 0.0f;
}

// ---------------------------------------------------------------- k1: W f32 -> bf16 (row-major + transposed)
__global__ __launch_bounds__(256) void convw_kernel(const float* __restrict__ W,
                                                    bf16* __restrict__ Wbf,
                                                    bf16* __restrict__ WbfT) {
  __shared__ bf16 tile[32][33];
  const int n0 = blockIdx.x * 32;
  const int d0 = blockIdx.y * 32;
  const int t = threadIdx.x;
#pragma unroll
  for (int i = 0; i < 4; ++i) {
    int idx = t + i * 256;
    int r = idx >> 5, c = idx & 31;   // d-local, n-local
    float w = W[(size_t)(d0 + r) * NN + (n0 + c)];
    bf16 h = (bf16)w;
    Wbf[(size_t)(d0 + r) * NN + (n0 + c)] = h;
    tile[c][r] = h;
  }
  __syncthreads();
#pragma unroll
  for (int i = 0; i < 4; ++i) {
    int idx = t + i * 256;
    int r = idx >> 5, c = idx & 31;   // n-local, d-local
    WbfT[(size_t)(n0 + r) * DD + (d0 + c)] = tile[r][c];
  }
}

// ---------------------------------------------------------------- k2: GEMM1  enc = tanh(dropout(x)·Wᵀ + b1)
// K-step 64; B tile via async global->LDS (overlaps with A dropout-convert).
__global__ __launch_bounds__(256) void gemm1_kernel(const float* __restrict__ x,
                                                    const float* __restrict__ mask,
                                                    const bf16* __restrict__ Wbf,
                                                    const float* __restrict__ b1,
                                                    float* __restrict__ encF,
                                                    bf16* __restrict__ encH) {
  __shared__ alignas(16) bf16 As[128 * S1];
  __shared__ alignas(16) bf16 Bs[128 * S1];
  const int t = threadIdx.x;
  const int lane = t & 31;
  const int wid = t >> 5;
  const int waveM = wid & 3;              // 4 waves * 32 rows (batch)
  const int waveN = wid >> 2;             // 2 waves * 64 cols (D)
  const int rowBase = blockIdx.y * 128;   // batch
  const int colBase = blockIdx.x * 128;   // D

  const v8f z8 = {0.f, 0.f, 0.f, 0.f, 0.f, 0.f, 0.f, 0.f};
  v8f acc[2][4];
#pragma unroll
  for (int i = 0; i < 2; ++i)
#pragma unroll
    for (int j = 0; j < 4; ++j) acc[i][j] = z8;

  // per-thread async B chunk constants (4 x 16B per 128x64 tile)
  uint32_t bLds[4], bOfs[4];
#pragma unroll
  for (int i = 0; i < 4; ++i) {
    int idx = t + i * 256;                // 0..1023
    int r = idx >> 3;
    int c8 = (idx & 7) << 3;
    bLds[i] = (uint32_t)(uintptr_t)(Bs + r * S1 + c8);
    bOfs[i] = (uint32_t)(((size_t)(colBase + r) * NN + c8) * sizeof(bf16));
  }
  const uint64_t baseB = (uint64_t)(uintptr_t)Wbf;

#pragma unroll 1
  for (int kt = 0; kt < NN / 64; ++kt) {
    const int k0 = kt * 64;
    // launch async B-tile copy (pure bf16 copy, no VALU)
#pragma unroll
    for (int i = 0; i < 4; ++i)
      async_cp16(bLds[i], bOfs[i] + (uint32_t)(k0 * sizeof(bf16)), baseB);

    // A tile: 128x64 f32 x*mask -> bf16 (overlaps with async B)
#pragma unroll
    for (int i = 0; i < 8; ++i) {
      int idx = t + i * 256;              // 0..2047
      int r = idx >> 4;
      int c4 = (idx & 15) << 2;
      const size_t g = (size_t)(rowBase + r) * NN + k0 + c4;
      const float4 xv = *(const float4*)(x + g);
      const float4 mv = *(const float4*)(mask + g);
      __builtin_prefetch(x + g + 64, 0, 1);
      v4bf h;
      h[0] = (bf16)(xv.x * mv.x * INV_KEEP);
      h[1] = (bf16)(xv.y * mv.y * INV_KEEP);
      h[2] = (bf16)(xv.z * mv.z * INV_KEEP);
      h[3] = (bf16)(xv.w * mv.w * INV_KEEP);
      *(v4bf*)(As + r * S1 + c4) = h;
    }
    wait_async_0();
    __syncthreads();
    mma_stage<S1>(As, Bs, 0, acc, waveM, waveN, lane);
    mma_stage<S1>(As, Bs, 32, acc, waveM, waveN, lane);
    __syncthreads();
  }

  // epilogue: + b1, tanh, store f32 + bf16
#pragma unroll
  for (int fj = 0; fj < 4; ++fj) {
    const int d = colBase + waveN * 64 + fj * 16 + (lane & 15);
    const float bias = b1[d];
#pragma unroll
    for (int fi = 0; fi < 2; ++fi) {
#pragma unroll
      for (int r = 0; r < 8; ++r) {
        const int b = rowBase + waveM * 32 + fi * 16 + r + ((lane & 16) ? 8 : 0);
        const float v = fast_tanh(acc[fi][fj][r] + bias);
        encF[(size_t)b * DD + d] = v;
        encH[(size_t)b * DD + d] = (bf16)v;
      }
    }
  }
}

// ---------------------------------------------------------------- k3: LayerNorm over D
__global__ __launch_bounds__(256) void ln_kernel(const float* __restrict__ encF,
                                                 const float* __restrict__ gamma,
                                                 const float* __restrict__ beta,
                                                 float* __restrict__ out) {
  const int b = blockIdx.x;
  const int t = threadIdx.x;
  const float4 v = ((const float4*)(encF + (size_t)b * DD))[t];
  float s = v.x + v.y + v.z + v.w;
  float q = v.x * v.x + v.y * v.y + v.z * v.z + v.w * v.w;
#pragma unroll
  for (int o = 16; o > 0; o >>= 1) { s += __shfl_down(s, o); q += __shfl_down(q, o); }
  __shared__ float ps[8], pq[8];
  if ((t & 31) == 0) { ps[t >> 5] = s; pq[t >> 5] = q; }
  __syncthreads();
  float S = 0.f, Q = 0.f;
#pragma unroll
  for (int i = 0; i < 8; ++i) { S += ps[i]; Q += pq[i]; }
  const float mu = S * (1.0f / DD);
  const float var = Q * (1.0f / DD) - mu * mu;
  const float rstd = rsqrtf(var + 1e-5f);
  const float4 g = ((const float4*)gamma)[t];
  const float4 be = ((const float4*)beta)[t];
  float4 o4;
  o4.x = (v.x - mu) * rstd * g.x + be.x;
  o4.y = (v.y - mu) * rstd * g.y + be.y;
  o4.z = (v.z - mu) * rstd * g.z + be.z;
  o4.w = (v.w - mu) * rstd * g.w + be.w;
  ((float4*)(out + (size_t)b * DD))[t] = o4;
}

// ---------------------------------------------------------------- k4: GEMM2 recon = enc·W + b2, fused loss
// Double-buffered async global->LDS pipeline (ASYNCcnt-throttled).
__global__ __launch_bounds__(256) void gemm2_kernel(const bf16* __restrict__ encH,
                                                    const bf16* __restrict__ WbfT,
                                                    const float* __restrict__ b2,
                                                    const float* __restrict__ x,
                                                    float* __restrict__ gErr,
                                                    float* __restrict__ gCnt) {
  __shared__ alignas(16) bf16 As[2][128 * S2];
  __shared__ alignas(16) bf16 Bs[2][128 * S2];
  __shared__ float rowErr[128];
  __shared__ float rowCnt[128];
  const int t = threadIdx.x;
  const int lane = t & 31;
  const int wid = t >> 5;
  const int waveM = wid & 3;
  const int waveN = wid >> 2;
  const int rowBase = blockIdx.y * 128;   // batch
  const int colBase = blockIdx.x * 128;   // N

  if (t < 128) { rowErr[t] = 0.f; rowCnt[t] = 0.f; }

  const v8f z8 = {0.f, 0.f, 0.f, 0.f, 0.f, 0.f, 0.f, 0.f};
  v8f acc[2][4];
#pragma unroll
  for (int i = 0; i < 2; ++i)
#pragma unroll
    for (int j = 0; j < 4; ++j) acc[i][j] = z8;

  // per-thread chunk constants: 2 x 16B per 128x32 tile
  int r_[2], c8_[2];
  uint32_t aOfs[2], bOfs[2];
#pragma unroll
  for (int i = 0; i < 2; ++i) {
    int idx = t + i * 256;                // 0..511
    r_[i] = idx >> 2;
    c8_[i] = (idx & 3) << 3;
    aOfs[i] = (uint32_t)(((size_t)(rowBase + r_[i]) * DD + c8_[i]) * sizeof(bf16));
    bOfs[i] = (uint32_t)(((size_t)(colBase + r_[i]) * DD + c8_[i]) * sizeof(bf16));
  }
  const uint64_t baseA = (uint64_t)(uintptr_t)encH;
  const uint64_t baseB = (uint64_t)(uintptr_t)WbfT;

#define ISSUE(BUF, K0)                                                            \
  {                                                                               \
    for (int i = 0; i < 2; ++i) {                                                 \
      uint32_t la = (uint32_t)(uintptr_t)(&As[BUF][r_[i] * S2 + c8_[i]]);         \
      uint32_t lb = (uint32_t)(uintptr_t)(&Bs[BUF][r_[i] * S2 + c8_[i]]);         \
      async_cp16(la, aOfs[i] + (uint32_t)((K0) * sizeof(bf16)), baseA);           \
      async_cp16(lb, bOfs[i] + (uint32_t)((K0) * sizeof(bf16)), baseB);           \
    }                                                                             \
  }

  ISSUE(0, 0);                            // prologue: stage kt=0 into buf0
#pragma unroll 1
  for (int kp = 0; kp < DD / 64; ++kp) {  // 16 double-steps over 32 K-tiles
    const int ktB = 2 * kp + 1;
    ISSUE(1, ktB * 32);                   // prefetch buf1 while buf0 computes
    wait_async_le4();                     // buf0 batch complete (in-order ASYNCcnt)
    __syncthreads();
    mma_stage<S2>(As[0], Bs[0], 0, acc, waveM, waveN, lane);
    __syncthreads();
    if (ktB + 1 < DD / 32) {
      ISSUE(0, (ktB + 1) * 32);           // prefetch buf0 while buf1 computes
      wait_async_le4();
    } else {
      wait_async_0();
    }
    __syncthreads();
    mma_stage<S2>(As[1], Bs[1], 0, acc, waveM, waveN, lane);
    __syncthreads();
  }
#undef ISSUE

  // fused loss epilogue: recon never touches HBM
#pragma unroll
  for (int fj = 0; fj < 4; ++fj) {
    const int n = colBase + waveN * 64 + fj * 16 + (lane & 15);
    const float bias = b2[n];
#pragma unroll
    for (int fi = 0; fi < 2; ++fi) {
#pragma unroll
      for (int r = 0; r < 8; ++r) {
        const int rl = waveM * 32 + fi * 16 + r + ((lane & 16) ? 8 : 0);
        const int b = rowBase + rl;
        const float recon = acc[fi][fj][r] + bias;
        const float xv = x[(size_t)b * NN + n];
        const float nz = (xv != 0.0f) ? 1.0f : 0.0f;
        const float e = nz * (xv - recon);
        atomicAdd(&rowErr[rl], e * e);
        atomicAdd(&rowCnt[rl], nz);
      }
    }
  }
  __syncthreads();
  if (t < 128) {
    atomicAdd(&gErr[rowBase + t], rowErr[t]);
    atomicAdd(&gCnt[rowBase + t], rowCnt[t]);
  }
}

// ---------------------------------------------------------------- k5: loss = sum_b err/cnt
__global__ __launch_bounds__(256) void loss_kernel(const float* __restrict__ gErr,
                                                   const float* __restrict__ gCnt,
                                                   float* __restrict__ out) {
  float s = 0.f;
  for (int b = threadIdx.x; b < BATCH; b += 256)
    s += gErr[b] / fmaxf(gCnt[b], 1.0f);
  __shared__ float sm[256];
  sm[threadIdx.x] = s;
  __syncthreads();
  for (int o = 128; o > 0; o >>= 1) {
    if (threadIdx.x < o) sm[threadIdx.x] += sm[threadIdx.x + o];
    __syncthreads();
  }
  if (threadIdx.x == 0) out[(size_t)BATCH * DD] = sm[0];
}

// ---------------------------------------------------------------- launch
extern "C" void kernel_launch(void* const* d_in, const int* in_sizes, int n_in,
                              void* d_out, int out_size, void* d_ws, size_t ws_size,
                              hipStream_t stream) {
  (void)in_sizes; (void)n_in; (void)out_size; (void)ws_size;
  const float* x    = (const float*)d_in[0];
  const float* mask = (const float*)d_in[1];
  const float* W    = (const float*)d_in[2];
  const float* b1   = (const float*)d_in[3];
  const float* b2   = (const float*)d_in[4];
  const float* gam  = (const float*)d_in[5];
  const float* bet  = (const float*)d_in[6];
  float* out = (float*)d_out;

  char* ws = (char*)d_ws;
  const size_t SZ_ENCF = (size_t)BATCH * DD * sizeof(float);   // 16 MB
  const size_t SZ_ENCH = (size_t)BATCH * DD * sizeof(bf16);    // 8 MB
  const size_t SZ_WBF  = (size_t)DD * NN * sizeof(bf16);       // 16 MB
  float* encF = (float*)ws;
  bf16*  encH = (bf16*)(ws + SZ_ENCF);
  bf16*  Wbf  = (bf16*)(ws + SZ_ENCF + SZ_ENCH);
  bf16*  WbfT = (bf16*)(ws + SZ_ENCF + SZ_ENCH + SZ_WBF);
  float* gErr = (float*)(ws + SZ_ENCF + SZ_ENCH + 2 * SZ_WBF);
  float* gCnt = gErr + BATCH;

  zero_kernel<<<(2 * BATCH + 255) / 256, 256, 0, stream>>>(gErr, 2 * BATCH);
  convw_kernel<<<dim3(NN / 32, DD / 32), 256, 0, stream>>>(W, Wbf, WbfT);
  gemm1_kernel<<<dim3(DD / 128, BATCH / 128), 256, 0, stream>>>(x, mask, Wbf, b1, encF, encH);
  ln_kernel<<<BATCH, 256, 0, stream>>>(encF, gam, bet, out);
  gemm2_kernel<<<dim3(NN / 128, BATCH / 128), 256, 0, stream>>>(encH, WbfT, b2, x, gErr, gCnt);
  loss_kernel<<<1, 256, 0, stream>>>(gErr, gCnt, out);
}
